// HMM_2336462209366
// MI455X (gfx1250) — compile-verified, matrix operand order checked by reading, not run on hardware
//
#include <hip/hip_runtime.h>
#include <hip/hip_bf16.h>
#include <cstddef>

// ---------------- problem constants ----------------
#define B_   8
#define T_   128
#define F_   80
#define N_   256
#define E_   256
#define H_   512
#define P_   256
#define G_   2048            // 4*H
#define CD_  161             // 2F+1
#define CDS_ 164             // Zin row stride, padded so rows stay 16B aligned
#define CDP_ 176             // hproj row stride (11 tiles of 16)
#define NEGV (-100000.0f)
#define LOG2PI_ 1.8378770664093453f

// ---------------- WMMA types ----------------
typedef __attribute__((ext_vector_type(16))) __bf16 v16bf;
typedef __attribute__((ext_vector_type(8)))  float  v8f;

__device__ inline v8f wmma_bf16(v16bf a, v16bf b, v8f c) {
  // D = A(16x32) * B(32x16) + C, fp32 accumulate
  return __builtin_amdgcn_wmma_f32_16x16x32_bf16(false, a, false, b, (short)0, c, false, false);
}

// float -> bf16 (round to nearest even), as raw u16
__device__ inline unsigned short f2bf(float x) {
  unsigned u = __float_as_uint(x);
  unsigned r = u + 0x7FFFu + ((u >> 16) & 1u);
  return (unsigned short)(r >> 16);
}
__device__ inline unsigned pack2(float lo, float hi) {
  return (unsigned)f2bf(lo) | ((unsigned)f2bf(hi) << 16);
}

// A-fragments are stored in LDS in per-lane fragment order:
// per K-chunk: 32 lanes x 8 u32 contiguous  ->  2x ds_load_b128 per fragment.
__device__ inline v16bf ldsA(const unsigned* pk, int kc, int lane) {
  union { v16bf v; uint4 q[2]; } f;
  const unsigned* p = pk + kc * 256 + lane * 8;
  f.q[0] = *(const uint4*)(p);
  f.q[1] = *(const uint4*)(p + 4);
  return f.v;
}

// Fragment element mapping for a pack-buffer flat index i (within one K-chunk set):
// i = kc*256 + lane*8 + v ; lane -> (m = lane&15, g = lane>>4) ;
// within-chunk k2 pair index = (v>>2)*8 + g*4 + (v&3) ; k = kc*32 + 2*k2 + h.
__device__ inline void frag_coords(int i, int& kc, int& m, int& k2l) {
  kc = i >> 8;
  int rem = i & 255, lane = rem >> 3, v = rem & 7;
  m = lane & 15;
  int g = lane >> 4;
  k2l = ((v >> 2) << 3) + (g << 2) + (v & 3);   // 0..15 within chunk
}

// fragment from global, pre-packed tile-major: 512 bf16 per tile, lane-major (2x b128/lane)
__device__ inline v16bf ldgB(const unsigned short* base, int tile, int lane) {
  return *reinterpret_cast<const v16bf*>(base + ((size_t)tile * 32 + lane) * 16);
}

// ---------------- math helpers ----------------
__device__ inline float softplusf_(float x) { return (x > 20.f) ? x : log1pf(expf(x)); }
__device__ inline float logsigf_(float x)   { return -softplusf_(-x); }
__device__ inline float sigmoidf_(float x)  { return 1.f / (1.f + expf(-x)); }
__device__ inline float logaddexpf_(float a, float b) {
  float m = fmaxf(a, b);
  return m + log1pf(expf(-fabsf(a - b)));
}
__device__ inline float emis1(float mel, float mean, float sz) {
  float stdv = softplusf_(sz) + 0.001f;
  float d = (mel - mean) / stdv;
  return -0.5f * d * d - logf(stdv);
}

// ---------------- shared memory layout (bytes, all offsets 64B-aligned) ----------------
#define SO_AR    0                      // 3*256 u32  (ar_t bf16 fragments, K padded to 96)
#define SO_XF    (SO_AR   + 3072)       // 16*256 f32 (prenet f32 staging)
#define SO_PK    (SO_XF   + 16384)      // 8*256 u32  (packed activation fragments, K<=256)
#define SO_HPK   (SO_PK   + 8192)       // 16*256 u32 (packed h fragments, K=512)
#define SO_BIAS  (SO_HPK  + 16384)      // 2048 f32   (b_ih + b_hh, staged once)
#define SO_H     (SO_BIAS + 8192)       // 8*512 f32
#define SO_C     (SO_H    + 16384)      // 8*512 f32
#define SO_G     (SO_C    + 16384)      // 8*2048 f32 (gates)
#define SO_HP    (SO_G    + 65536)      // 8*176 f32  (h projection, 16B-aligned rows)
#define SO_MEL   (SO_HP   + 5632)       // 8*80 f32
#define SO_EM    (SO_MEL  + 2560)       // 8*256 f32
#define SO_TV    (SO_EM   + 8192)       // 8*256 f32
#define SO_LA    (SO_TV   + 8192)       // 8*256 f32
#define SO_RED   (SO_LA   + 8192)       // 8*256 f32
#define SO_MAX   (SO_RED  + 8192)       // 16 f32 (8 used: holds log_c per batch)
#define SO_LLA   (SO_MAX  + 64)         // 8*256 f32
#define SO_LTV   (SO_LLA  + 8192)       // 8*256 f32
#define SO_LOGC  (SO_LTV  + 8192)       // 8*128 f32
#define SMEM_TOTAL (SO_LOGC + 4096)     // ~207 KB (<= 320 KB/WG per CDNA5 ISA)

// per-b logsumexp over 256 states; v0 -> idx tid (b 0..3), v1 -> idx tid+1024 (b 4..7)
// On return sMax[b] = logsumexp_n. All 1024 threads must call.
__device__ inline void lse8x256(float* sRed, float* sMax, float v0, float v1, int tid) {
  const int i0 = tid, i1 = tid + 1024;
  sRed[i0] = v0; sRed[i1] = v1;
#pragma unroll
  for (int s = 128; s > 0; s >>= 1) {
    __syncthreads();
    { int n = i0 & 255; if (n < s) { float* p = sRed + (i0 & ~255); p[n] = fmaxf(p[n], p[n + s]); } }
    { int n = i1 & 255; if (n < s) { float* p = sRed + (i1 & ~255); p[n] = fmaxf(p[n], p[n + s]); } }
  }
  __syncthreads();
  float m0 = sRed[i0 & ~255], m1 = sRed[i1 & ~255];
  __syncthreads();
  sRed[i0] = expf(v0 - m0); sRed[i1] = expf(v1 - m1);
#pragma unroll
  for (int s = 128; s > 0; s >>= 1) {
    __syncthreads();
    { int n = i0 & 255; if (n < s) { float* p = sRed + (i0 & ~255); p[n] += p[n + s]; } }
    { int n = i1 & 255; if (n < s) { float* p = sRed + (i1 & ~255); p[n] += p[n + s]; } }
  }
  __syncthreads();
  if ((i0 & 255) == 0) sMax[i0 >> 8] = m0 + logf(sRed[i0]);
  if ((i1 & 255) == 0) sMax[i1 >> 8] = m1 + logf(sRed[i1]);
  __syncthreads();
}

// ---------------- weight packing: fragment order for WMMA ----------------
// tile-major (nt,kc); within a tile: lane-major 16 bf16 (two b128 per lane)
__global__ void pack_tiles(const float* __restrict__ src, unsigned short* __restrict__ dst,
                           int Kreal, int Nreal, int ld, int trans, int kchunks, int total) {
  int id = blockIdx.x * blockDim.x + threadIdx.x;
  if (id >= total) return;
  int tile = id >> 9, w = id & 511;
  int lane = w >> 4, e = w & 15, v = e >> 1, h = e & 1;
  int nt = tile / kchunks, kc = tile % kchunks;
  int n = nt * 16 + (lane & 15);
  int k = kc * 32 + ((v >> 2) << 4) + ((lane >> 4) << 3) + ((v & 3) << 1) + h;
  float val = 0.f;
  if (k < Kreal && n < Nreal)
    val = trans ? src[(size_t)n * ld + k] : src[(size_t)k * ld + n];
  dst[id] = f2bf(val);
}

// ---------------- Zin = inputs @ W_out[H:,:] + b_out  (time-invariant precompute) ----------------
__global__ __launch_bounds__(256) void zin_gemm(const unsigned short* __restrict__ inA,
                                                const unsigned short* __restrict__ woutE,
                                                const float* __restrict__ b_out,
                                                float* __restrict__ zin) {
  int tid = threadIdx.x, lane = tid & 31;
  int wid = blockIdx.x * 8 + (tid >> 5);
  if (wid >= 128 * 11) return;
  int mt = wid / 11, nt = wid % 11;
  v8f acc = {0.f, 0.f, 0.f, 0.f, 0.f, 0.f, 0.f, 0.f};
#pragma unroll
  for (int kc = 0; kc < 8; ++kc)
    acc = wmma_bf16(ldgB(inA, mt * 8 + kc, lane), ldgB(woutE, nt * 8 + kc, lane), acc);
  int n = nt * 16 + (lane & 15), g = lane >> 4;
  if (n < CD_) {
    float bo = b_out[n];
#pragma unroll
    for (int r = 0; r < 8; ++r) {
      int row = mt * 16 + r + g * 8;
      zin[(size_t)row * CDS_ + n] = acc[r] + bo;
    }
  }
}

// ---------------- persistent sequential scan (one workgroup, LDS-resident state) ----------------
__global__ __launch_bounds__(1024) void hmm_scan(
    const float* __restrict__ mels, const float* __restrict__ zin,
    const unsigned short* __restrict__ wpre1, const unsigned short* __restrict__ wpre2,
    const unsigned short* __restrict__ wih,   const unsigned short* __restrict__ whh,
    const unsigned short* __restrict__ wouth,
    const float* __restrict__ b_ih, const float* __restrict__ b_hh,
    const int* __restrict__ ilen, const int* __restrict__ mlen,
    float* __restrict__ out) {
  extern __shared__ __align__(16) unsigned char smem[];
  unsigned* sAr  = (unsigned*)(smem + SO_AR);
  float*    sXf  = (float*)(smem + SO_XF);
  unsigned* sPk  = (unsigned*)(smem + SO_PK);
  unsigned* sHpk = (unsigned*)(smem + SO_HPK);
  float*    sBias= (float*)(smem + SO_BIAS);
  float*    sH   = (float*)(smem + SO_H);
  float*    sC   = (float*)(smem + SO_C);
  float*    sG   = (float*)(smem + SO_G);
  float*    sHP  = (float*)(smem + SO_HP);
  float*    sMel = (float*)(smem + SO_MEL);
  float*    sEm  = (float*)(smem + SO_EM);
  float*    sTv  = (float*)(smem + SO_TV);
  float*    sLa  = (float*)(smem + SO_LA);
  float*    sRed = (float*)(smem + SO_RED);
  float*    sMax = (float*)(smem + SO_MAX);
  float*    sLLA = (float*)(smem + SO_LLA);
  float*    sLTV = (float*)(smem + SO_LTV);
  float*    sLogc= (float*)(smem + SO_LOGC);

  const int tid = threadIdx.x, lane = tid & 31, wave = tid >> 5;
  const int b0 = tid >> 8, b1 = (tid + 1024) >> 8;
  const int il0 = ilen[b0], il1 = ilen[b1];
  const int ml0 = mlen[b0], ml1 = mlen[b1];

  // initial state: h = c = 0, packed h = 0, staged gate bias
  for (int i = tid; i < B_ * H_; i += 1024) { sH[i] = 0.f; sC[i] = 0.f; }
  for (int i = tid; i < 16 * 256; i += 1024) sHpk[i] = 0u;
  for (int i = tid; i < B_ * N_; i += 1024) sLa[i] = NEGV;
  for (int i = tid; i < G_; i += 1024) sBias[i] = b_ih[i] + b_hh[i];
  __syncthreads();

  for (int t = 0; t < T_; ++t) {
    // ---- stage ar_t (= t==0 ? 0 : mels[:,t-1]) as bf16 A-fragments (16 x 96) and mel_t
    for (int i = tid; i < 3 * 256; i += 1024) {
      int kc, m, k2l; frag_coords(i, kc, m, k2l);
      int k = kc * 32 + 2 * k2l;
      float lo = 0.f, hi = 0.f;
      if (m < B_ && t > 0) {
        const float* row = mels + ((size_t)m * T_ + (t - 1)) * F_;
        if (k < F_)     lo = row[k];
        if (k + 1 < F_) hi = row[k + 1];
      }
      sAr[i] = pack2(lo, hi);
    }
    for (int i = tid; i < B_ * (F_ / 4); i += 1024) {   // mel_t, float4 loads
      int b = i / (F_ / 4), f4 = i % (F_ / 4);
      *(float4*)(sMel + b * F_ + 4 * f4) =
          *(const float4*)(mels + ((size_t)b * T_ + t) * F_ + 4 * f4);
    }
    if (t + 1 < T_ && tid < B_)
      __builtin_prefetch(mels + ((size_t)tid * T_ + (t + 1)) * F_, 0, 1); // global_prefetch_b8
    __syncthreads();

    // ---- prenet layer 1: relu((16x80) @ W_pre1) -> sXf
    if (wave < 16) {
      v8f acc = {0.f, 0.f, 0.f, 0.f, 0.f, 0.f, 0.f, 0.f};
#pragma unroll
      for (int kc = 0; kc < 3; ++kc)
        acc = wmma_bf16(ldsA(sAr, kc, lane), ldgB(wpre1, wave * 3 + kc, lane), acc);
      int n = wave * 16 + (lane & 15), g = lane >> 4;
#pragma unroll
      for (int r = 0; r < 8; ++r) sXf[(r + g * 8) * P_ + n] = fmaxf(acc[r], 0.f);
    }
    __syncthreads();
    for (int i = tid; i < 8 * 256; i += 1024) {          // pack P1 -> sPk fragments
      int kc, m, k2l; frag_coords(i, kc, m, k2l);
      int k = kc * 32 + 2 * k2l;
      sPk[i] = pack2(sXf[m * P_ + k], sXf[m * P_ + k + 1]);
    }
    __syncthreads();

    // ---- prenet layer 2: relu((16x256) @ W_pre2) -> sXf
    if (wave < 16) {
      v8f acc = {0.f, 0.f, 0.f, 0.f, 0.f, 0.f, 0.f, 0.f};
#pragma unroll
      for (int kc = 0; kc < 8; ++kc)
        acc = wmma_bf16(ldsA(sPk, kc, lane), ldgB(wpre2, wave * 8 + kc, lane), acc);
      int n = wave * 16 + (lane & 15), g = lane >> 4;
#pragma unroll
      for (int r = 0; r < 8; ++r) sXf[(r + g * 8) * P_ + n] = fmaxf(acc[r], 0.f);
    }
    __syncthreads();
    for (int i = tid; i < 8 * 256; i += 1024) {          // pack P -> sPk fragments
      int kc, m, k2l; frag_coords(i, kc, m, k2l);
      int k = kc * 32 + 2 * k2l;
      sPk[i] = pack2(sXf[m * P_ + k], sXf[m * P_ + k + 1]);
    }
    __syncthreads();

    // ---- LSTM gates: p @ W_ih^T + h @ W_hh^T + biases  (16 x 2048, 128 N-tiles / 32 waves)
#pragma unroll
    for (int ti = 0; ti < 4; ++ti) {
      int nt = wave * 4 + ti;
      v8f acc = {0.f, 0.f, 0.f, 0.f, 0.f, 0.f, 0.f, 0.f};
#pragma unroll
      for (int kc = 0; kc < 8; ++kc)
        acc = wmma_bf16(ldsA(sPk, kc, lane), ldgB(wih, nt * 8 + kc, lane), acc);
#pragma unroll
      for (int kc = 0; kc < 16; ++kc)
        acc = wmma_bf16(ldsA(sHpk, kc, lane), ldgB(whh, nt * 16 + kc, lane), acc);
      int n = nt * 16 + (lane & 15), g = lane >> 4;
      float bias = sBias[n];
#pragma unroll
      for (int r = 0; r < 8; ++r) {
        int m = r + g * 8;
        if (m < B_) sG[m * G_ + n] = acc[r] + bias;
      }
    }
    __syncthreads();

    // ---- LSTM cell elementwise
    for (int i = tid; i < B_ * H_; i += 1024) {
      int b = i >> 9, j = i & 511;
      float ig = sigmoidf_(sG[b * G_ + j]);
      float fg = sigmoidf_(sG[b * G_ + H_ + j]);
      float gg = tanhf(sG[b * G_ + 2 * H_ + j]);
      float og = sigmoidf_(sG[b * G_ + 3 * H_ + j]);
      float cn = fg * sC[i] + ig * gg;
      sC[i] = cn;
      sH[i] = og * tanhf(cn);
    }
    __syncthreads();
    for (int i = tid; i < 16 * 256; i += 1024) {         // pack h -> sHpk fragments
      int kc, m, k2l; frag_coords(i, kc, m, k2l);
      int k = kc * 32 + 2 * k2l;
      sHpk[i] = (m < B_) ? pack2(sH[m * H_ + k], sH[m * H_ + k + 1]) : 0u;
    }
    __syncthreads();

    // ---- h projection: h @ W_out[:H]  (8 x 161, 11 N-tiles)
    if (wave < 11) {
      v8f acc = {0.f, 0.f, 0.f, 0.f, 0.f, 0.f, 0.f, 0.f};
#pragma unroll
      for (int kc = 0; kc < 16; ++kc)
        acc = wmma_bf16(ldsA(sHpk, kc, lane), ldgB(wouth, wave * 16 + kc, lane), acc);
      int n = wave * 16 + (lane & 15), g = lane >> 4;
#pragma unroll
      for (int r = 0; r < 8; ++r) {
        int m = r + g * 8;
        if (m < B_) sHP[m * CDP_ + n] = acc[r];
      }
    }
    __syncthreads();

    // ---- Gaussian emission + transition logit, per (b,n): z = Zin[b,n] + hproj[b]
#pragma unroll
    for (int q = 0; q < 2; ++q) {
      int idx = tid + q * 1024, b = idx >> 8, n = idx & 255;
      const float* z  = zin + (size_t)(b * N_ + n) * CDS_;
      const float* hp = sHP + b * CDP_;
      const float* ml = sMel + b * F_;
      const int il = q ? il1 : il0;
      float acc = 0.f;
      if (n < il) {
#pragma unroll 5
        for (int f = 0; f < F_; f += 4) {          // all rows 16B-aligned -> b128 loads
          float4 zm = *(const float4*)(z + f);
          float4 zs = *(const float4*)(z + F_ + f);
          float4 hm = *(const float4*)(hp + f);
          float4 hs = *(const float4*)(hp + F_ + f);
          float4 mm = *(const float4*)(ml + f);
          acc += emis1(mm.x, zm.x + hm.x, zs.x + hs.x);
          acc += emis1(mm.y, zm.y + hm.y, zs.y + hs.y);
          acc += emis1(mm.z, zm.z + hm.z, zs.z + hs.z);
          acc += emis1(mm.w, zm.w + hm.w, zs.w + hs.w);
        }
        acc -= 0.5f * LOG2PI_ * (float)F_;
      }
      sEm[idx] = acc;
      sTv[idx] = z[2 * F_] + hp[2 * F_];
    }
    __syncthreads();

    // ---- HMM forward recursion
    float lat[2];
#pragma unroll
    for (int q = 0; q < 2; ++q) {
      int idx = tid + q * 1024, n = idx & 255;
      const int il = q ? il1 : il0;
      float tvc = sTv[idx];
      float staying = sLa[idx] + logsigf_(-tvc);
      float leaving = (n == 0) ? NEGV : (sLa[idx - 1] + logsigf_(sTv[idx - 1]));
      float trans = (n < il) ? logaddexpf_(staying, leaving) : NEGV;
      lat[q] = (t == 0) ? (((n == 0) ? 0.f : NEGV) + sEm[idx]) : (sEm[idx] + trans);
    }
    lse8x256(sRed, sMax, lat[0], lat[1], tid);   // sMax[b] = log_c_t
#pragma unroll
    for (int q = 0; q < 2; ++q) {
      int idx = tid + q * 1024, b = idx >> 8;
      float las = lat[q] - sMax[b];
      sLa[idx] = las;
      if (t == (q ? ml1 : ml0) - 1) { sLLA[idx] = las; sLTV[idx] = sTv[idx]; }
    }
    if (tid < B_) sLogc[tid * T_ + t] = sMax[tid];
    __syncthreads();
  }

  // ---- finalize: sum(log_c masked) + logsumexp_n(last_la + log_trans)
  float v0, v1;
  {
    int n = tid & 255;
    float la = (n < il0) ? sLLA[tid] : NEGV;
    float lt = (n == il0 - 1) ? logsigf_(sLTV[tid]) : NEGV;
    v0 = la + lt;
  }
  {
    int idx = tid + 1024, n = idx & 255;
    float la = (n < il1) ? sLLA[idx] : NEGV;
    float lt = (n == il1 - 1) ? logsigf_(sLTV[idx]) : NEGV;
    v1 = la + lt;
  }
  lse8x256(sRed, sMax, v0, v1, tid);
  if (tid < B_) {
    float s = 0.f;
    int L = mlen[tid];
    for (int tt = 0; tt < L; ++tt) s += sLogc[tid * T_ + tt];
    out[tid] = s + sMax[tid];
  }
}

// ---------------- host ----------------
extern "C" void kernel_launch(void* const* d_in, const int* in_sizes, int n_in,
                              void* d_out, int out_size, void* d_ws, size_t ws_size,
                              hipStream_t stream) {
  const float* inputs     = (const float*)d_in[0];   // (8,256,256)
  const float* mels       = (const float*)d_in[1];   // (8,128,80)
  const float* W_pre1     = (const float*)d_in[2];   // (80,256)
  const float* W_pre2     = (const float*)d_in[3];   // (256,256)
  const float* W_ih       = (const float*)d_in[4];   // (2048,256)
  const float* W_hh       = (const float*)d_in[5];   // (2048,512)
  const float* b_ih       = (const float*)d_in[6];   // (2048,)
  const float* b_hh       = (const float*)d_in[7];   // (2048,)
  const float* W_out      = (const float*)d_in[8];   // (768,161)
  const float* b_out      = (const float*)d_in[9];   // (161,)
  const int*   inputs_len = (const int*)d_in[10];    // (8,)
  const int*   mel_lens   = (const int*)d_in[11];    // (8,)
  float* out = (float*)d_out;

  // workspace carve (bf16 fragment-packed weights, then fp32 Zin) — ~6 MB total
  unsigned short* ws16 = (unsigned short*)d_ws;
  size_t o = 0;
  unsigned short* wpre1 = ws16 + o; o += (size_t)16 * 3 * 512;
  unsigned short* wpre2 = ws16 + o; o += (size_t)16 * 8 * 512;
  unsigned short* wih   = ws16 + o; o += (size_t)128 * 8 * 512;
  unsigned short* whh   = ws16 + o; o += (size_t)128 * 16 * 512;
  unsigned short* wouth = ws16 + o; o += (size_t)11 * 16 * 512;
  unsigned short* woutE = ws16 + o; o += (size_t)11 * 8 * 512;
  unsigned short* inA   = ws16 + o; o += (size_t)128 * 8 * 512;
  float* zin = (float*)(ws16 + o);                   // (2048, CDS_=164) f32, 16B-aligned rows

  // pack weights / inputs into WMMA fragment order
  // (src, dst, Kreal, Nreal, ld, trans, kchunks, total)
  {
    int tot;
    tot = 16 * 3 * 512;
    pack_tiles<<<(tot + 255) / 256, 256, 0, stream>>>(W_pre1, wpre1, 80, 256, 256, 0, 3, tot);
    tot = 16 * 8 * 512;
    pack_tiles<<<(tot + 255) / 256, 256, 0, stream>>>(W_pre2, wpre2, 256, 256, 256, 0, 8, tot);
    tot = 128 * 8 * 512;
    pack_tiles<<<(tot + 255) / 256, 256, 0, stream>>>(W_ih, wih, 256, 2048, 256, 1, 8, tot);
    tot = 128 * 16 * 512;
    pack_tiles<<<(tot + 255) / 256, 256, 0, stream>>>(W_hh, whh, 512, 2048, 512, 1, 16, tot);
    tot = 11 * 16 * 512;
    pack_tiles<<<(tot + 255) / 256, 256, 0, stream>>>(W_out, wouth, 512, 161, 161, 0, 16, tot);
    tot = 11 * 8 * 512;
    pack_tiles<<<(tot + 255) / 256, 256, 0, stream>>>(W_out + (size_t)H_ * CD_, woutE, 256, 161, 161, 0, 8, tot);
    tot = 128 * 8 * 512;
    pack_tiles<<<(tot + 255) / 256, 256, 0, stream>>>(inputs, inA, 256, 2048, 256, 1, 8, tot);
  }

  // time-invariant Zin = inputs @ W_out[H:] + b_out   (128 x 11 tiles, 8 waves/block)
  zin_gemm<<<176, 256, 0, stream>>>(inA, woutE, b_out, zin);

  // persistent sequential scan: 1 workgroup, 1024 threads, ~207 KB dynamic LDS
  hmm_scan<<<1, 1024, SMEM_TOTAL, stream>>>(mels, zin, wpre1, wpre2, wih, whh, wouth,
                                            b_ih, b_hh, inputs_len, mel_lens, out);
}
